// GINDeepSigns_64665027609009
// MI455X (gfx1250) — compile-verified, matrix operand order checked by reading, not run on hardware
//
#include <hip/hip_runtime.h>
#include <hip/hip_bf16.h>
#include <cstdint>

// ---------------------------------------------------------------------------
// GINDeepSigns forward for MI455X (gfx1250, wave32, WMMA bf16 16x16x32)
//   h: [N, 8, 64]  (8 sign-channels)   weights: 64x64 (+64->16 for rho)
//   Scatter-add runs on global f32 atomics (h fits in 192MB L2).
//   MLP GEMMs run on v_wmma_f32_16x16x32_bf16, weights staged transposed
//   in LDS so A and B fragments are contiguous 32-bit LDS (b128) loads.
//   GEMM epilogue restages C fragments through LDS -> coalesced b128 stores.
// ---------------------------------------------------------------------------

typedef __bf16 v16bf __attribute__((ext_vector_type(16)));
typedef float  v8f   __attribute__((ext_vector_type(8)));

#define LDW 72   // padded row stride (elements) for LDS bf16 tiles (even!)

__device__ __forceinline__ __bf16 f2bf(float f) {
    unsigned u = __float_as_uint(f);
    unsigned r = u + 0x7FFFu + ((u >> 16) & 1u);   // round-to-nearest-even
    unsigned short h = (unsigned short)(r >> 16);
    __bf16 out;
    __builtin_memcpy(&out, &h, 2);
    return out;
}

union FragU { v16bf bf; unsigned u[8]; };

// Load one 16x32 bf16 A/B fragment from an LDS tile laid out row-major
// [*, LDW].  Per the CDNA5 ISA layout (CDNA5 ISA §7.12.2, 16-bit A 16x32):
//   lane L: row/col = L&15 ; lanes 0-15 hold K = k0+0..7 and k0+16..23,
//   lanes 16-31 hold K = k0+8..15 and k0+24..31 (2 elems per VGPR).
__device__ __forceinline__ v16bf load_frag(const __bf16* base, int row, int k0, int lane) {
    int kh = ((lane >> 4) & 1) << 3;
    const unsigned* p32 = (const unsigned*)(base + row * LDW + k0 + kh); // 4B aligned
    FragU f;
#pragma unroll
    for (int i = 0; i < 4; ++i) f.u[i] = p32[i];
#pragma unroll
    for (int i = 0; i < 4; ++i) f.u[4 + i] = p32[8 + i];   // +16 elements
    return f.bf;
}

// One wave computes D[16 x NT*16] = A[16x64] * BT^T + bias, optional ReLU.
// A    : LDS bf16 tile [16][LDW]
// BT   : LDS bf16 transposed weights [NT*16][LDW]  (BT[n][k] = W[k][n])
// out  : per-lane C fragments; element r maps to M = r + 8*(lane>>4),
//        N = nt*16 + (lane&15)   (CDNA5 ISA 32-bit C/D layout)
template <int NT>
__device__ __forceinline__ void wave_gemm(const __bf16* A, const __bf16* BT,
                                          const float* bias, bool relu, int lane,
                                          float (&out)[NT][8]) {
    v8f acc[NT];
#pragma unroll
    for (int nt = 0; nt < NT; ++nt) acc[nt] = {};
    int rc = lane & 15;
#pragma unroll
    for (int kb = 0; kb < 2; ++kb) {
        v16bf a = load_frag(A, rc, kb * 32, lane);
#pragma unroll
        for (int nt = 0; nt < NT; ++nt) {
            v16bf b = load_frag(BT, nt * 16 + rc, kb * 32, lane);
            acc[nt] = __builtin_amdgcn_wmma_f32_16x16x32_bf16(
                false, a, false, b, (short)0, acc[nt], false, false);
        }
    }
    int n = lane & 15;
#pragma unroll
    for (int nt = 0; nt < NT; ++nt)
#pragma unroll
        for (int r = 0; r < 8; ++r) {
            float v = acc[nt][r] + bias[nt * 16 + n];
            out[nt][r] = (relu && v < 0.f) ? 0.f : v;
        }
}

// Restage per-lane C fragments into a per-wave LDS float buffer [16][NT*16]
// (row-major, stride NT*16), then cooperatively store coalesced float4s.
template <int NT>
__device__ __forceinline__ void store_tile(float* __restrict__ F /*LDS*/,
                                           float* __restrict__ out /*global*/,
                                           const float (&v)[NT][8],
                                           int row0, int rows, int lane) {
    const int W = NT * 16;                 // floats per row
    int n = lane & 15, mo = (lane >> 4) * 8;
#pragma unroll
    for (int nt = 0; nt < NT; ++nt)
#pragma unroll
        for (int r = 0; r < 8; ++r)
            F[(mo + r) * W + nt * 16 + n] = v[nt][r];
    // wave-local LDS is in-order: safe to read back without a block barrier
    const int Q = W / 4;                   // float4s per row
    for (int i = lane; i < 16 * Q; i += 32) {
        int m = i / Q, q = i % Q;
        int row = row0 + m;
        if (row < rows)
            ((float4*)(out + (size_t)row * W))[q] = ((const float4*)F)[m * Q + q];
    }
}

// ---------------------------------------------------------------------------
// Layer-0 scalar aggregation: s[n][c] = sign_c * x[n][c&3] + sum_{e: dst=n} ...
// ---------------------------------------------------------------------------
__global__ void init_s_kernel(const float* __restrict__ x, float* __restrict__ s, int N) {
    int t = blockIdx.x * blockDim.x + threadIdx.x;
    if (t >= N * 8) return;
    int n = t >> 3, c = t & 7;
    float v = x[n * 4 + (c & 3)];
    s[t] = (c < 4) ? v : -v;
}

__global__ void scatter0_kernel(const float* __restrict__ x, const long long* __restrict__ ei,
                                float* __restrict__ s, int E) {
    int e = blockIdx.x * blockDim.x + threadIdx.x;
    if (e >= E) return;
    int src = (int)ei[e];
    int dst = (int)ei[E + e];
    float4 v = *(const float4*)(x + (size_t)src * 4);
    float* sd = s + (size_t)dst * 8;
    atomicAdd(sd + 0, v.x);  atomicAdd(sd + 1, v.y);
    atomicAdd(sd + 2, v.z);  atomicAdd(sd + 3, v.w);
    atomicAdd(sd + 4, -v.x); atomicAdd(sd + 5, -v.y);
    atomicAdd(sd + 6, -v.z); atomicAdd(sd + 7, -v.w);
}

// ---------------------------------------------------------------------------
// Layer-0 MLP: in = scalar s per row; t = relu(s*w1 + b1) [64]; out = t@W2+b2
// ---------------------------------------------------------------------------
__global__ void mlp0_kernel(const float* __restrict__ s,
                            const float* __restrict__ w1, const float* __restrict__ b1,
                            const float* __restrict__ W2, const float* __restrict__ b2,
                            float* __restrict__ out, int rows) {
    __shared__ __align__(16) __bf16 sW2t[64 * LDW];
    __shared__ float sw1[64], sb1[64], sb2[64];
    __shared__ __align__(16) __bf16 sT[8][16 * LDW];

    int tid = threadIdx.x;
    for (int i = tid; i < 64 * 64; i += 256) {
        int k = i >> 6, n = i & 63;
        sW2t[n * LDW + k] = f2bf(W2[i]);
    }
    if (tid < 64) { sw1[tid] = w1[tid]; sb1[tid] = b1[tid]; sb2[tid] = b2[tid]; }
    __syncthreads();

    int wave = tid >> 5, lane = tid & 31;
    int row0 = (blockIdx.x * 8 + wave) * 16;
    int rmax = rows - 1;
    __bf16* T = sT[wave];
    for (int i = lane; i < 16 * 64; i += 32) {
        int m = i >> 6, j = i & 63;
        int r = row0 + m; r = r > rmax ? rmax : r;
        float v = s[r] * sw1[j] + sb1[j];
        T[m * LDW + j] = f2bf(v > 0.f ? v : 0.f);
    }
    float o[4][8];
    wave_gemm<4>(T, sW2t, sb2, false, lane, o);
    // T fully consumed by the GEMM -> reuse its LDS as f32 output staging
    store_tile<4>((float*)T, out, o, row0, rows, lane);
}

// ---------------------------------------------------------------------------
// Vector scatter: hagg[dst*8+c][0:64] += h[src*8+c][0:64]  (f32 L2 atomics)
// ---------------------------------------------------------------------------
__global__ void scatter_vec_kernel(const float* __restrict__ h, float* __restrict__ hagg,
                                   const long long* __restrict__ ei, int E) {
    int t = blockIdx.x * blockDim.x + threadIdx.x;
    if (t >= E * 8) return;
    int e = t >> 3, c = t & 7;
    int src = (int)ei[e], dst = (int)ei[E + e];
    const float4* hs = (const float4*)(h + ((size_t)src * 8 + c) * 64);
    float* hd = hagg + ((size_t)dst * 8 + c) * 64;
    __builtin_prefetch(hs, 0, 3);
#pragma unroll
    for (int q = 0; q < 16; ++q) {
        float4 v = hs[q];
        atomicAdd(hd + q * 4 + 0, v.x);
        atomicAdd(hd + q * 4 + 1, v.y);
        atomicAdd(hd + q * 4 + 2, v.z);
        atomicAdd(hd + q * 4 + 3, v.w);
    }
}

// ---------------------------------------------------------------------------
// Generic 64->64(relu)->64 MLP over rows, WMMA bf16, wave = 16-row strip
// ---------------------------------------------------------------------------
__global__ void mlp_pair_kernel(const float* __restrict__ in, float* __restrict__ out,
                                const float* __restrict__ W1, const float* __restrict__ b1,
                                const float* __restrict__ W2, const float* __restrict__ b2,
                                int rows) {
    __shared__ __align__(16) __bf16 sW1t[64 * LDW], sW2t[64 * LDW];
    __shared__ float sb1[64], sb2[64];
    __shared__ __align__(16) __bf16 sA[8][16 * LDW];
    __shared__ __align__(16) __bf16 sT[8][16 * LDW];

    int tid = threadIdx.x;
    for (int i = tid; i < 64 * 64; i += 256) {
        int k = i >> 6, n = i & 63;
        sW1t[n * LDW + k] = f2bf(W1[i]);
        sW2t[n * LDW + k] = f2bf(W2[i]);
    }
    if (tid < 64) { sb1[tid] = b1[tid]; sb2[tid] = b2[tid]; }
    __syncthreads();

    int wave = tid >> 5, lane = tid & 31;
    int row0 = (blockIdx.x * 8 + wave) * 16;
    int rmax = rows - 1;
    __bf16* A = sA[wave];
    __bf16* T = sT[wave];

    // stage 16 rows x 64 f32 -> bf16 LDS (float4 loads, 8 per lane)
    for (int i = lane; i < 256; i += 32) {
        int m = i >> 4, q = i & 15;
        int r = row0 + m; r = r > rmax ? rmax : r;
        float4 v = ((const float4*)(in + (size_t)r * 64))[q];
        int e = m * LDW + q * 4;
        A[e + 0] = f2bf(v.x); A[e + 1] = f2bf(v.y);
        A[e + 2] = f2bf(v.z); A[e + 3] = f2bf(v.w);
    }

    float t4[4][8];
    wave_gemm<4>(A, sW1t, sb1, true, lane, t4);   // layer 1 + ReLU
    int n = lane & 15, mo = (lane >> 4) * 8;
#pragma unroll
    for (int nt = 0; nt < 4; ++nt)
#pragma unroll
        for (int r = 0; r < 8; ++r)
            T[(mo + r) * LDW + nt * 16 + n] = f2bf(t4[nt][r]);

    float o4[4][8];
    wave_gemm<4>(T, sW2t, sb2, false, lane, o4);  // layer 2
    // A-tile fully consumed -> reuse as f32 output staging
    store_tile<4>((float*)A, out, o4, row0, rows, lane);
}

// ---------------------------------------------------------------------------
// Sign-invariant reduce: hsum[n][k] = 0.25 * sum_{c=0..7} h[n][c][k]
// ---------------------------------------------------------------------------
__global__ void reduce_kernel(const float* __restrict__ h, float* __restrict__ hsum, int N) {
    int t = blockIdx.x * blockDim.x + threadIdx.x;
    if (t >= N * 64) return;
    int n = t >> 6, k = t & 63;
    const float* p = h + ((size_t)n * 8) * 64 + k;
    float a = 0.f;
#pragma unroll
    for (int c = 0; c < 8; ++c) a += p[c * 64];
    hsum[t] = 0.25f * a;
}

// ---------------------------------------------------------------------------
// rho MLP: 64 -> 64 (relu) -> 16
// ---------------------------------------------------------------------------
__global__ void rho_kernel(const float* __restrict__ in, float* __restrict__ out,
                           const float* __restrict__ W1, const float* __restrict__ b1,
                           const float* __restrict__ W2, const float* __restrict__ b2,
                           int rows) {
    __shared__ __align__(16) __bf16 sW1t[64 * LDW], sW2t[16 * LDW];
    __shared__ float sb1[64], sb2[16];
    __shared__ __align__(16) __bf16 sA[8][16 * LDW];
    __shared__ __align__(16) __bf16 sT[8][16 * LDW];

    int tid = threadIdx.x;
    for (int i = tid; i < 64 * 64; i += 256) {
        int k = i >> 6, n = i & 63;
        sW1t[n * LDW + k] = f2bf(W1[i]);
    }
    for (int i = tid; i < 64 * 16; i += 256) {
        int k = i >> 4, n = i & 15;
        sW2t[n * LDW + k] = f2bf(W2[i]);
    }
    if (tid < 64) sb1[tid] = b1[tid];
    if (tid < 16) sb2[tid] = b2[tid];
    __syncthreads();

    int wave = tid >> 5, lane = tid & 31;
    int row0 = (blockIdx.x * 8 + wave) * 16;
    int rmax = rows - 1;
    __bf16* A = sA[wave];
    __bf16* T = sT[wave];

    for (int i = lane; i < 256; i += 32) {
        int m = i >> 4, q = i & 15;
        int r = row0 + m; r = r > rmax ? rmax : r;
        float4 v = ((const float4*)(in + (size_t)r * 64))[q];
        int e = m * LDW + q * 4;
        A[e + 0] = f2bf(v.x); A[e + 1] = f2bf(v.y);
        A[e + 2] = f2bf(v.z); A[e + 3] = f2bf(v.w);
    }

    float t4[4][8];
    wave_gemm<4>(A, sW1t, sb1, true, lane, t4);
    int n = lane & 15, mo = (lane >> 4) * 8;
#pragma unroll
    for (int nt = 0; nt < 4; ++nt)
#pragma unroll
        for (int r = 0; r < 8; ++r)
            T[(mo + r) * LDW + nt * 16 + n] = f2bf(t4[nt][r]);

    float o1[1][8];
    wave_gemm<1>(T, sW2t, sb2, false, lane, o1);
    // A-tile fully consumed -> reuse as f32 output staging ([16][16] floats)
    store_tile<1>((float*)A, out, o1, row0, rows, lane);
}

// ---------------------------------------------------------------------------
extern "C" void kernel_launch(void* const* d_in, const int* in_sizes, int n_in,
                              void* d_out, int out_size, void* d_ws, size_t ws_size,
                              hipStream_t stream) {
    const float* x        = (const float*)d_in[0];
    const long long* ei   = (const long long*)d_in[1];
    const float* c0_W1 = (const float*)d_in[2];
    const float* c0_b1 = (const float*)d_in[3];
    const float* c0_W2 = (const float*)d_in[4];
    const float* c0_b2 = (const float*)d_in[5];
    const float* c1_W1 = (const float*)d_in[6];
    const float* c1_b1 = (const float*)d_in[7];
    const float* c1_W2 = (const float*)d_in[8];
    const float* c1_b2 = (const float*)d_in[9];
    const float* c2_W1 = (const float*)d_in[10];
    const float* c2_b1 = (const float*)d_in[11];
    const float* c2_W2 = (const float*)d_in[12];
    const float* c2_b2 = (const float*)d_in[13];
    const float* r_W1  = (const float*)d_in[14];
    const float* r_b1  = (const float*)d_in[15];
    const float* r_W2  = (const float*)d_in[16];
    const float* r_b2  = (const float*)d_in[17];
    float* outp = (float*)d_out;

    const int N = in_sizes[0] / 4;
    const int E = in_sizes[1] / 2;
    const int rows8 = N * 8;
    const size_t hsz = (size_t)rows8 * 64;

    float* s    = (float*)d_ws;
    float* hA   = s + (((size_t)rows8 + 63) & ~(size_t)63);
    float* hB   = hA + hsz;
    float* hsum = hB + hsz;

    const int B = 256;
    int g_init  = (rows8 + B - 1) / B;
    int g_edge  = (E + B - 1) / B;
    int g_sc    = (E * 8 + B - 1) / B;
    int g_mlp   = (rows8 + 127) / 128;
    int g_red   = (N * 64 + B - 1) / B;
    int g_rho   = (N + 127) / 128;

    // ---- layer 0 (scalar channel) ----
    init_s_kernel<<<g_init, B, 0, stream>>>(x, s, N);
    scatter0_kernel<<<g_edge, B, 0, stream>>>(x, ei, s, E);
    mlp0_kernel<<<g_mlp, B, 0, stream>>>(s, c0_W1, c0_b1, c0_W2, c0_b2, hA, rows8);

    // ---- layer 1 ----
    hipMemcpyAsync(hB, hA, hsz * sizeof(float), hipMemcpyDeviceToDevice, stream);
    scatter_vec_kernel<<<g_sc, B, 0, stream>>>(hA, hB, ei, E);
    mlp_pair_kernel<<<g_mlp, B, 0, stream>>>(hB, hA, c1_W1, c1_b1, c1_W2, c1_b2, rows8);

    // ---- layer 2 ----
    hipMemcpyAsync(hB, hA, hsz * sizeof(float), hipMemcpyDeviceToDevice, stream);
    scatter_vec_kernel<<<g_sc, B, 0, stream>>>(hA, hB, ei, E);
    mlp_pair_kernel<<<g_mlp, B, 0, stream>>>(hB, hA, c2_W1, c2_b1, c2_W2, c2_b2, rows8);

    // ---- sign-invariant reduce + rho ----
    reduce_kernel<<<g_red, B, 0, stream>>>(hA, hsum, N);
    rho_kernel<<<g_rho, B, 0, stream>>>(hsum, outp, r_W1, r_b1, r_W2, r_b2, N);
}
